// SlidingWindowAttention_88562225643873
// MI455X (gfx1250) — matrix operand order repaired and checked by reading, not compiled, not run on hardware
//
#include <hip/hip_runtime.h>
#include <stdint.h>

// ---------------------------------------------------------------------------
// Sliding-window attention block for MI455X (gfx1250, wave32, WMMA).
//   EMBED=1024, HEADS=16, HEAD_DIM=64, lookback HALF=512, B=2, T=4096.
// All matmuls on v_wmma_f32_16x16x32_bf16 (bf16 in, f32 accum).
// Workspace: Q[B,H,T,D] bf16, K[B,H,T,D] bf16, V^T[B,H,D,T] bf16,
//            O_attn[B,T,C] bf16  -> 4 x 16MB = 64MB, L2-resident.
// ---------------------------------------------------------------------------

#define EMBED    1024
#define HEADS    16
#define HEAD_DIM 64
#define HALF_WIN 512
#define SEQ      4096
#define BATCH    2
#define MROWS    (BATCH * SEQ)     /* 8192 */
#define QK_SCALE 0.125f            /* 64^-0.5 */

typedef __attribute__((ext_vector_type(16))) __bf16 v16bf;
typedef __attribute__((ext_vector_type(8)))  __bf16 v8bf;
typedef __attribute__((ext_vector_type(4)))  __bf16 v4bf;
typedef __attribute__((ext_vector_type(8)))  float  v8f;

typedef unsigned short bf16_raw;   // host-safe handle for workspace pointers

// Native f32 -> bf16 (clang lowers with RNE semantics; gfx1250 has hardware
// cvt support, replacing the ~5-VALU software round sequence).
__device__ __forceinline__ __bf16 f2bf(float f) { return (__bf16)f; }

__device__ __forceinline__ v8f vzero() {
  v8f z;
#pragma unroll
  for (int i = 0; i < 8; ++i) z[i] = 0.0f;
  return z;
}

__device__ __forceinline__ v8f wmma_bf16(v16bf a, v16bf b, v8f c) {
  // (neg_a, A, neg_b, B, c_mod, C, reuse_a, reuse_b)
  return __builtin_amdgcn_wmma_f32_16x16x32_bf16(false, a, false, b,
                                                 (short)0, c, false, false);
}

// A-matrix fragment (16x32, M x K), row-major source with leading dim `ld`.
// Layout used here (contiguous K): lane l holds row M=(l&15); elems 0..15 = K (16g..16g+15),
//             g = l>>4.
__device__ __forceinline__ v16bf load_A(const __bf16* p, int ld, int row0, int col0) {
  const int l = threadIdx.x & 31, r = l & 15, g = l >> 4;
  const __bf16* base = p + (size_t)(row0 + r) * ld + col0 + 16 * g;
  v8bf lo = *(const v8bf*)(base);
  v8bf hi = *(const v8bf*)(base + 8);
  v16bf a;
#pragma unroll
  for (int i = 0; i < 8; ++i) { a[i] = lo[i]; a[8 + i] = hi[i]; }
  return a;
}

// B-matrix fragment (32 x 16, K x N) from "B^T" storage: storage row n holds
// column n of B contiguously over K (leading dim `ld`).
// Layout used here (contiguous K): lane l holds column N=(l&15); elems 0..15 = K (16g..16g+15).
__device__ __forceinline__ v16bf load_Bt(const __bf16* p, int ld, int n0, int k0) {
  const int l = threadIdx.x & 31, n = l & 15, g = l >> 4;
  return *(const v16bf*)(p + (size_t)(n0 + n) * ld + k0 + 16 * g);
}

// ---------------------------------------------------------------------------
// Kernel 1: fused QKV projections.  out[m,n] = sum_c X[m,c]*W[n,c] + b[n]
// grid.z picks (X, W, b): 0 -> Q (scaled), 1 -> K, 2 -> V (stored transposed)
// Block tile 128(M) x 128(N), BK=32, 8 waves each computing 32x64.
// Double-buffered LDS; global loads register-staged to overlap with WMMA.
// ---------------------------------------------------------------------------
__global__ __launch_bounds__(256) void qkv_proj_kernel(
    const float* __restrict__ q_in, const float* __restrict__ k_in,
    const float* __restrict__ v_in,
    const float* __restrict__ wq, const float* __restrict__ bq,
    const float* __restrict__ wk, const float* __restrict__ bk,
    const float* __restrict__ wv, const float* __restrict__ bv,
    bf16_raw* __restrict__ Qws_, bf16_raw* __restrict__ Kws_,
    bf16_raw* __restrict__ Vtws_)
{
  const int mode = blockIdx.z;
  const float* X    = (mode == 0) ? q_in : (mode == 1) ? k_in : v_in;
  const float* W    = (mode == 0) ? wq   : (mode == 1) ? wk   : wv;
  const float* bias = (mode == 0) ? bq   : (mode == 1) ? bk   : bv;
  __bf16* Qws  = (__bf16*)Qws_;
  __bf16* Kws  = (__bf16*)Kws_;
  __bf16* Vtws = (__bf16*)Vtws_;

  const int m0  = blockIdx.x * 128;
  const int n0  = blockIdx.y * 128;
  const int tid = threadIdx.x;

  __shared__ __bf16 sX[2][128 * 32];
  __shared__ __bf16 sW[2][128 * 32];

  const int w  = tid >> 5;
  const int wm = (w & 3) * 32;   // wave M offset within tile
  const int wn = (w >> 2) * 64;  // wave N offset within tile

  v8f acc[2][4];
#pragma unroll
  for (int i = 0; i < 2; ++i)
#pragma unroll
    for (int j = 0; j < 4; ++j) acc[i][j] = vzero();

  float4 rx[4], rw[4];                 // register staging for next K tile
  auto loadregs = [&](int k0) {
#pragma unroll
    for (int j = 0; j < 4; ++j) {
      int idx = tid + 256 * j;         // 0..1023 float4 chunks
      int row = idx >> 3;
      int c4  = (idx & 7) << 2;
      rx[j] = *(const float4*)(X + (size_t)(m0 + row) * EMBED + k0 + c4);
      rw[j] = *(const float4*)(W + (size_t)(n0 + row) * EMBED + k0 + c4);
    }
  };
  auto store_stage = [&](__bf16* bx, __bf16* bw) {
#pragma unroll
    for (int j = 0; j < 4; ++j) {
      int idx = tid + 256 * j;
      int row = idx >> 3;
      int c4  = (idx & 7) << 2;
      v4bf tx; tx[0] = f2bf(rx[j].x); tx[1] = f2bf(rx[j].y);
               tx[2] = f2bf(rx[j].z); tx[3] = f2bf(rx[j].w);
      *(v4bf*)(bx + row * 32 + c4) = tx;
      v4bf tw; tw[0] = f2bf(rw[j].x); tw[1] = f2bf(rw[j].y);
               tw[2] = f2bf(rw[j].z); tw[3] = f2bf(rw[j].w);
      *(v4bf*)(bw + row * 32 + c4) = tw;
    }
  };

  loadregs(0);
  int cur = 0;
  for (int k0 = 0; k0 < EMBED; k0 += 32) {
    store_stage(&sX[cur][0], &sW[cur][0]);
    __syncthreads();
    if (k0 + 32 < EMBED) loadregs(k0 + 32);   // in flight during compute

    v16bf a0 = load_A(&sX[cur][0], 32, wm, 0);
    v16bf a1 = load_A(&sX[cur][0], 32, wm + 16, 0);
#pragma unroll
    for (int j = 0; j < 4; ++j) {
      v16bf bb = load_Bt(&sW[cur][0], 32, wn + j * 16, 0);
      acc[0][j] = wmma_bf16(a0, bb, acc[0][j]);
      acc[1][j] = wmma_bf16(a1, bb, acc[1][j]);
    }
    cur ^= 1;
  }

  // Epilogue.  C layout: elem r, lane l -> row (r + 8*(l>>4)), col (l&15).
  const int l = tid & 31, lr = l & 15, g = l >> 4;
  const float sc = (mode == 0) ? QK_SCALE : 1.0f;
#pragma unroll
  for (int i = 0; i < 2; ++i) {
#pragma unroll
    for (int j = 0; j < 4; ++j) {
      const int n  = n0 + wn + j * 16 + lr;
      const float bn = bias[n];
      const int h = n >> 6, d = n & 63;
      if (mode < 2) {
        __bf16* dst = (mode == 0) ? Qws : Kws;
#pragma unroll
        for (int r = 0; r < 8; ++r) {
          int m = m0 + wm + 16 * i + 8 * g + r;
          int b = m >> 12, t = m & (SEQ - 1);
          dst[((size_t)(b * HEADS + h) * SEQ + t) * HEAD_DIM + d] =
              f2bf((acc[i][j][r] + bn) * sc);
        }
      } else {
        // V stored transposed [B,H,D,T]: 8 consecutive t per lane -> 16B store
        int mb = m0 + wm + 16 * i + 8 * g;
        int b = mb >> 12, t = mb & (SEQ - 1);
        v8bf ov;
#pragma unroll
        for (int r = 0; r < 8; ++r) ov[r] = f2bf(acc[i][j][r] + bn);
        *(v8bf*)(Vtws + ((size_t)(b * HEADS + h) * HEAD_DIM + d) * SEQ + t) = ov;
      }
    }
  }
}

// ---------------------------------------------------------------------------
// Kernel 2: sliding-window flash attention.
// 1 wave per (32-query block, h, b): two 16-row A tiles share every K/V
// B-fragment -> 16 WMMA per 32-key block against 8 fragment loads.
// Keys j valid for query i iff  i-512 <= j <= i.
// ---------------------------------------------------------------------------
__global__ __launch_bounds__(32) void swa_kernel(
    const bf16_raw* __restrict__ Qws_, const bf16_raw* __restrict__ Kws_,
    const bf16_raw* __restrict__ Vtws_, bf16_raw* __restrict__ Ows_)
{
  const int qs = blockIdx.x * 32;
  const int h  = blockIdx.y;
  const int b  = blockIdx.z;
  const int l = threadIdx.x & 31, lr = l & 15, g = l >> 4;

  const __bf16* Qp = (const __bf16*)Qws_  + (size_t)(b * HEADS + h) * SEQ * HEAD_DIM;
  const __bf16* Kp = (const __bf16*)Kws_  + (size_t)(b * HEADS + h) * SEQ * HEAD_DIM;
  const __bf16* Vt = (const __bf16*)Vtws_ + (size_t)(b * HEADS + h) * HEAD_DIM * SEQ;
  __bf16* Ows = (__bf16*)Ows_;

  __shared__ __bf16 Plds[2][16 * 32];   // per-q-tile P staging (C -> A layout)

  v16bf qa[2][2];
#pragma unroll
  for (int qt = 0; qt < 2; ++qt) {
    qa[qt][0] = load_A(Qp, HEAD_DIM, qs + 16 * qt, 0);
    qa[qt][1] = load_A(Qp, HEAD_DIM, qs + 16 * qt, 32);
  }

  float mrow[2][8], lrow[2][8];
#pragma unroll
  for (int qt = 0; qt < 2; ++qt)
#pragma unroll
    for (int r = 0; r < 8; ++r) { mrow[qt][r] = -1e30f; lrow[qt][r] = 0.0f; }
  v8f acc[2][4];
#pragma unroll
  for (int qt = 0; qt < 2; ++qt)
#pragma unroll
    for (int dt = 0; dt < 4; ++dt) acc[qt][dt] = vzero();

  int start = qs - HALF_WIN; if (start < 0) start = 0;
  start &= ~31;

  for (int kb = start; kb <= qs + 31; kb += 32) {
    // shared B fragments for this 32-key block
    const v16bf kf00 = load_Bt(Kp, HEAD_DIM, kb,       0);
    const v16bf kf01 = load_Bt(Kp, HEAD_DIM, kb,       32);
    const v16bf kf10 = load_Bt(Kp, HEAD_DIM, kb + 16,  0);
    const v16bf kf11 = load_Bt(Kp, HEAD_DIM, kb + 16,  32);
    v16bf vf[4];
#pragma unroll
    for (int dt = 0; dt < 4; ++dt) vf[dt] = load_Bt(Vt, SEQ, dt * 16, kb);

#pragma unroll
    for (int qt = 0; qt < 2; ++qt) {
      v8f s0 = vzero(), s1 = vzero();
      s0 = wmma_bf16(qa[qt][0], kf00, s0);
      s0 = wmma_bf16(qa[qt][1], kf01, s0);
      s1 = wmma_bf16(qa[qt][0], kf10, s1);
      s1 = wmma_bf16(qa[qt][1], kf11, s1);

      float p0[8], p1[8], corr[8];
#pragma unroll
      for (int r = 0; r < 8; ++r) {
        const int i  = qs + 16 * qt + 8 * g + r;
        const int j0 = kb + lr, j1 = kb + 16 + lr;
        float v0 = ((j0 <= i) && (j0 >= i - HALF_WIN)) ? s0[r] : -1e30f;
        float v1 = ((j1 <= i) && (j1 >= i - HALF_WIN)) ? s1[r] : -1e30f;
        // row max across the 16 lanes holding this row (xor masks < 16)
        float mx = fmaxf(v0, v1);
        mx = fmaxf(mx, __shfl_xor(mx, 1, 32));
        mx = fmaxf(mx, __shfl_xor(mx, 2, 32));
        mx = fmaxf(mx, __shfl_xor(mx, 4, 32));
        mx = fmaxf(mx, __shfl_xor(mx, 8, 32));
        const float mn = fmaxf(mrow[qt][r], mx);
        corr[r] = __expf(mrow[qt][r] - mn);
        p0[r] = __expf(v0 - mn);
        p1[r] = __expf(v1 - mn);
        float sum = p0[r] + p1[r];
        sum += __shfl_xor(sum, 1, 32);
        sum += __shfl_xor(sum, 2, 32);
        sum += __shfl_xor(sum, 4, 32);
        sum += __shfl_xor(sum, 8, 32);
        lrow[qt][r] = lrow[qt][r] * corr[r] + sum;
        mrow[qt][r] = mn;
      }
#pragma unroll
      for (int dt = 0; dt < 4; ++dt)
#pragma unroll
        for (int r = 0; r < 8; ++r) acc[qt][dt][r] *= corr[r];

      // P: C-layout -> LDS [m][k_local]
#pragma unroll
      for (int r = 0; r < 8; ++r) {
        const int m = 8 * g + r;
        Plds[qt][m * 32 + lr]      = f2bf(p0[r]);
        Plds[qt][m * 32 + 16 + lr] = f2bf(p1[r]);
      }
    }
    __syncthreads();
#pragma unroll
    for (int qt = 0; qt < 2; ++qt) {
      const v16bf pa = load_A(&Plds[qt][0], 32, 0, 0);
#pragma unroll
      for (int dt = 0; dt < 4; ++dt)
        acc[qt][dt] = wmma_bf16(pa, vf[dt], acc[qt][dt]);
    }
    __syncthreads();
  }

  // normalize and store O_attn[b, t, h*64 + d]
#pragma unroll
  for (int qt = 0; qt < 2; ++qt) {
    float inv[8];
#pragma unroll
    for (int r = 0; r < 8; ++r) inv[r] = 1.0f / lrow[qt][r];
#pragma unroll
    for (int dt = 0; dt < 4; ++dt)
#pragma unroll
      for (int r = 0; r < 8; ++r) {
        const int t = qs + 16 * qt + 8 * g + r;
        const int c = h * HEAD_DIM + dt * 16 + lr;
        Ows[((size_t)b * SEQ + t) * EMBED + c] = f2bf(acc[qt][dt][r] * inv[r]);
      }
  }
}

// ---------------------------------------------------------------------------
// Kernel 3: output projection.  out[m,n] = sum_c O[m,c]*Wo[n,c] + bo[n] (f32)
// Same double-buffered structure as kernel 1; A tile already bf16.
// ---------------------------------------------------------------------------
__global__ __launch_bounds__(256) void out_proj_kernel(
    const bf16_raw* __restrict__ A_, const float* __restrict__ W,
    const float* __restrict__ bias, float* __restrict__ out)
{
  const __bf16* A = (const __bf16*)A_;
  const int m0  = blockIdx.x * 128;
  const int n0  = blockIdx.y * 128;
  const int tid = threadIdx.x;

  __shared__ __bf16 sX[2][128 * 32];
  __shared__ __bf16 sW[2][128 * 32];

  const int w  = tid >> 5;
  const int wm = (w & 3) * 32;
  const int wn = (w >> 2) * 64;

  v8f acc[2][4];
#pragma unroll
  for (int i = 0; i < 2; ++i)
#pragma unroll
    for (int j = 0; j < 4; ++j) acc[i][j] = vzero();

  v8bf ra[2];
  float4 rw[4];
  auto loadregs = [&](int k0) {
#pragma unroll
    for (int j = 0; j < 2; ++j) {            // A tile: 512 8-elem chunks
      int idx = tid + 256 * j;
      int row = idx >> 2;
      int c8  = (idx & 3) << 3;
      ra[j] = *(const v8bf*)(A + (size_t)(m0 + row) * EMBED + k0 + c8);
    }
#pragma unroll
    for (int j = 0; j < 4; ++j) {            // W tile: f32
      int idx = tid + 256 * j;
      int row = idx >> 3;
      int c4  = (idx & 7) << 2;
      rw[j] = *(const float4*)(W + (size_t)(n0 + row) * EMBED + k0 + c4);
    }
  };
  auto store_stage = [&](__bf16* bx, __bf16* bw) {
#pragma unroll
    for (int j = 0; j < 2; ++j) {
      int idx = tid + 256 * j;
      int row = idx >> 2;
      int c8  = (idx & 3) << 3;
      *(v8bf*)(bx + row * 32 + c8) = ra[j];
    }
#pragma unroll
    for (int j = 0; j < 4; ++j) {
      int idx = tid + 256 * j;
      int row = idx >> 3;
      int c4  = (idx & 7) << 2;
      v4bf tw; tw[0] = f2bf(rw[j].x); tw[1] = f2bf(rw[j].y);
               tw[2] = f2bf(rw[j].z); tw[3] = f2bf(rw[j].w);
      *(v4bf*)(bw + row * 32 + c4) = tw;
    }
  };

  loadregs(0);
  int cur = 0;
  for (int k0 = 0; k0 < EMBED; k0 += 32) {
    store_stage(&sX[cur][0], &sW[cur][0]);
    __syncthreads();
    if (k0 + 32 < EMBED) loadregs(k0 + 32);

    v16bf a0 = load_A(&sX[cur][0], 32, wm, 0);
    v16bf a1 = load_A(&sX[cur][0], 32, wm + 16, 0);
#pragma unroll
    for (int j = 0; j < 4; ++j) {
      v16bf bb = load_Bt(&sW[cur][0], 32, wn + j * 16, 0);
      acc[0][j] = wmma_bf16(a0, bb, acc[0][j]);
      acc[1][j] = wmma_bf16(a1, bb, acc[1][j]);
    }
    cur ^= 1;
  }

  const int l = tid & 31, lr = l & 15, g = l >> 4;
#pragma unroll
  for (int i = 0; i < 2; ++i) {
#pragma unroll
    for (int j = 0; j < 4; ++j) {
      const int n = n0 + wn + j * 16 + lr;
      const float bn = bias[n];
#pragma unroll
      for (int r = 0; r < 8; ++r) {
        int m = m0 + wm + 16 * i + 8 * g + r;
        out[(size_t)m * EMBED + n] = acc[i][j][r] + bn;
      }
    }
  }
}

// ---------------------------------------------------------------------------
extern "C" void kernel_launch(void* const* d_in, const int* in_sizes, int n_in,
                              void* d_out, int out_size, void* d_ws, size_t ws_size,
                              hipStream_t stream)
{
  const float* query = (const float*)d_in[0];
  const float* key_  = (const float*)d_in[1];
  const float* value = (const float*)d_in[2];
  const float* wq = (const float*)d_in[3];
  const float* bq = (const float*)d_in[4];
  const float* wk = (const float*)d_in[5];
  const float* bk = (const float*)d_in[6];
  const float* wv = (const float*)d_in[7];
  const float* bv = (const float*)d_in[8];
  const float* wo = (const float*)d_in[9];
  const float* bo = (const float*)d_in[10];
  float* out = (float*)d_out;

  const size_t TEN = (size_t)MROWS * EMBED;   // 8M bf16 elements per tensor
  bf16_raw* Qws  = (bf16_raw*)d_ws;
  bf16_raw* Kws  = Qws + TEN;
  bf16_raw* Vtws = Kws + TEN;
  bf16_raw* Ows  = Vtws + TEN;                // total 64 MB of workspace

  // 1) Q/K/V projections (z = 0/1/2)
  dim3 gproj(MROWS / 128, EMBED / 128, 3);
  qkv_proj_kernel<<<gproj, 256, 0, stream>>>(
      query, key_, value, wq, bq, wk, bk, wv, bv, Qws, Kws, Vtws);

  // 2) sliding-window attention (32 queries per wave)
  dim3 gattn(SEQ / 32, HEADS, BATCH);
  swa_kernel<<<gattn, 32, 0, stream>>>(Qws, Kws, Vtws, Ows);

  // 3) output projection
  dim3 gout(MROWS / 128, EMBED / 128, 1);
  out_proj_kernel<<<gout, 256, 0, stream>>>(Ows, wo, bo, out);
}
